// MultiHeadAttention_64312840290713
// MI455X (gfx1250) — matrix operand order repaired and checked by reading, not compiled, not run on hardware
//
#include <hip/hip_runtime.h>
#include <math.h>

typedef __attribute__((ext_vector_type(4)))  _Float16 v4h;
typedef __attribute__((ext_vector_type(8)))  _Float16 v8h;
typedef __attribute__((ext_vector_type(16))) _Float16 v16h;
typedef __attribute__((ext_vector_type(8)))  float    v8f;
typedef __attribute__((ext_vector_type(4)))  unsigned int v4u;
typedef __attribute__((ext_vector_type(8)))  int      v8i;
typedef __attribute__((ext_vector_type(4)))  int      v4i;

#define T_SEQ   32
#define C_DIM   128
#define N_HEADS 4
#define HDIM    32
#define WPITCH  136   // halves: 128-wide f16 rows, padded (272B = 68 dwords -> conflict-free)
#define SPITCH  40    // halves: 32-wide f16 rows, padded (80B)
#define WEIP    33    // dwords: 32-wide f32 rows, padded
#define WAVES_PER_BLOCK 8
#define ATT_SCALE 11.313708498984760f   // sqrt(128)

#define WEIGHT_LDS_BYTES (4 * 128 * WPITCH * 2)   // 139264
#define WAVE_LDS_BYTES   20608
#define TOTAL_LDS_BYTES  (WEIGHT_LDS_BYTES + WAVES_PER_BLOCK * WAVE_LDS_BYTES) // 304128

__device__ __forceinline__ v8f wmma_f16(v16h a, v16h b, v8f c) {
  return __builtin_amdgcn_wmma_f32_16x16x32_f16(false, a, false, b, (short)0, c,
                                                false, false);
}

__device__ __forceinline__ v8f zero8() {
  v8f z = {0.f, 0.f, 0.f, 0.f, 0.f, 0.f, 0.f, 0.f};
  return z;
}

// A-matrix 16x32 f16 fragment built straight from global f32 (row-major, stride 128).
__device__ __forceinline__ v16h a_frag_global_f32(const float* rowbase, int k0) {
  const int lane = threadIdx.x & 31;
  const float* p = rowbase + (lane & 15) * C_DIM + k0 + ((lane >> 4) << 3);
  float4 f0 = *(const float4*)(p);
  float4 f1 = *(const float4*)(p + 4);
  float4 f2 = *(const float4*)(p + 16);
  float4 f3 = *(const float4*)(p + 20);
  v16h a;
  a[0]  = (_Float16)f0.x; a[1]  = (_Float16)f0.y; a[2]  = (_Float16)f0.z; a[3]  = (_Float16)f0.w;
  a[4]  = (_Float16)f1.x; a[5]  = (_Float16)f1.y; a[6]  = (_Float16)f1.z; a[7]  = (_Float16)f1.w;
  a[8]  = (_Float16)f2.x; a[9]  = (_Float16)f2.y; a[10] = (_Float16)f2.z; a[11] = (_Float16)f2.w;
  a[12] = (_Float16)f3.x; a[13] = (_Float16)f3.y; a[14] = (_Float16)f3.z; a[15] = (_Float16)f3.w;
  return a;
}

// A fragment from row-major f16 LDS tile (base pre-offset to the 16-row m-tile).
__device__ __forceinline__ v16h a_frag_lds(const _Float16* base, int pitch, int k0) {
  const int lane = threadIdx.x & 31;
  const _Float16* p = base + (lane & 15) * pitch + k0 + ((lane >> 4) << 3);
  v8h lo = *(const v8h*)(p);
  v8h hi = *(const v8h*)(p + 16);
  return __builtin_shufflevector(lo, hi, 0, 1, 2, 3, 4, 5, 6, 7,
                                 8, 9, 10, 11, 12, 13, 14, 15);
}

// B-matrix 32x16 f16 fragment: source stored N-major [n][k] (row n contiguous in k).
__device__ __forceinline__ v16h b_frag_lds(const _Float16* base, int pitch) {
  const int lane = threadIdx.x & 31;
  const _Float16* p = base + (lane & 15) * pitch + ((lane >> 4) << 3);
  v8h lo = *(const v8h*)(p);
  v8h hi = *(const v8h*)(p + 16);
  return __builtin_shufflevector(lo, hi, 0, 1, 2, 3, 4, 5, 6, 7,
                                 8, 9, 10, 11, 12, 13, 14, 15);
}

// D-tile (v8f, lane holds column n=L&15, rows m = (L>>4)*8 + i) stores:
__device__ __forceinline__ void d_store_rowmajor_h(_Float16* base, int pitch, v8f d) {
  const int lane = threadIdx.x & 31;
  _Float16* p = base + ((lane >> 4) << 3) * pitch + (lane & 15);
#pragma unroll
  for (int i = 0; i < 8; ++i) p[i * pitch] = (_Float16)d[i];
}

__device__ __forceinline__ void d_store_colmajor_h(_Float16* base, int pitch, v8f d) {
  const int lane = threadIdx.x & 31;
  v8h h;
#pragma unroll
  for (int i = 0; i < 8; ++i) h[i] = (_Float16)d[i];
  *(v8h*)(base + (lane & 15) * pitch + ((lane >> 4) << 3)) = h;
}

__device__ __forceinline__ void d_store_rowmajor_f(float* base, int pitch, v8f d) {
  const int lane = threadIdx.x & 31;
  float* p = base + ((lane >> 4) << 3) * pitch + (lane & 15);
#pragma unroll
  for (int i = 0; i < 8; ++i) p[i * pitch] = d[i];
}

__device__ __forceinline__ void lds_order() { asm volatile("" ::: "memory"); }

// ---------------------------------------------------------------------------
// Pre-kernel: convert the four 128x128 f32 weight matrices to f16 in d_ws with
// the padded WPITCH layout (exactly the image the TDM copies into LDS).
// 16384 threads, one float4 each.
// ---------------------------------------------------------------------------
__global__ void __launch_bounds__(256, 1)
cvt_weights_kernel(const float* __restrict__ Wq, const float* __restrict__ Wk,
                   const float* __restrict__ Wv, const float* __restrict__ Wp,
                   _Float16* __restrict__ ws) {
  const int i = blockIdx.x * 256 + threadIdx.x;   // 0..16383 (float4 groups)
  const int w  = i >> 12;                         // matrix index 0..3
  const int e  = (i & 4095) << 2;                 // element within matrix
  const int r  = e >> 7;
  const int c  = e & 127;
  const float* s = (w == 0) ? Wq : (w == 1) ? Wk : (w == 2) ? Wv : Wp;
  float4 f = *(const float4*)(s + e);
  v4h h;
  h[0] = (_Float16)f.x; h[1] = (_Float16)f.y;
  h[2] = (_Float16)f.z; h[3] = (_Float16)f.w;
  *(v4h*)(ws + (size_t)w * 128 * WPITCH + r * WPITCH + c) = h;
}

// ---------------------------------------------------------------------------
// Main kernel
// ---------------------------------------------------------------------------
__global__ void __launch_bounds__(256, 1)
mha_wmma_kernel(const float* __restrict__ x,  const _Float16* __restrict__ ws,
                const float* __restrict__ bp, float* __restrict__ out, int nB) {
  extern __shared__ char smem[];
  _Float16* wq_s = (_Float16*)smem;
  _Float16* wk_s = wq_s + 128 * WPITCH;
  _Float16* wv_s = wk_s + 128 * WPITCH;
  _Float16* wp_s = wv_s + 128 * WPITCH;

  const int tid  = threadIdx.x;
  const int wave = tid >> 5;
  const int lane = tid & 31;

  // ---- TDM: one tensor_load_to_lds pulls all f16 weights (139264 B) into LDS ----
  if (wave == 0) {
    const unsigned long long gaddr = (unsigned long long)ws;
    const unsigned int       laddr = (unsigned int)(unsigned long long)(void*)smem;
    v4u g0;
    g0[0] = 1u;                                     // count=1, user mode
    g0[1] = laddr;                                  // lds_addr [63:32]
    g0[2] = (unsigned int)gaddr;                    // global_addr [95:64]
    g0[3] = (unsigned int)((gaddr >> 32) & 0x1FFFFFFu) | (2u << 30); // addr hi + type=2
    v8i g1;
    g1[0] = (1 << 16);      // workgroup_mask=0, data_size=1 (2 bytes)
    g1[1] = (WPITCH << 16); // tensor_dim0 = 136 (low 16)
    g1[2] = (512 << 16);    // tensor_dim0 hi=0, tensor_dim1 = 512 rows (low 16)
    g1[3] = (WPITCH << 16); // tensor_dim1 hi=0, tile_dim0 = 136
    g1[4] = 512;            // tile_dim1 = 512, tile_dim2 = 0
    g1[5] = WPITCH;         // tensor_dim0_stride = 136 (low 32)
    g1[6] = 0;              // dim0_stride hi, dim1_stride lo
    g1[7] = 0;              // dim1_stride hi
    v4i g2 = {0, 0, 0, 0};
    v4i g3 = {0, 0, 0, 0};
    v8i g4 = {0, 0, 0, 0, 0, 0, 0, 0};
    __builtin_amdgcn_tensor_load_to_lds(g0, g1, g2, g3, g4, 0);
    __builtin_amdgcn_s_wait_tensorcnt(0);
  }
  __syncthreads();

  const int b = blockIdx.x * WAVES_PER_BLOCK + wave;
  if (b < nB) {
    char* wbase = smem + WEIGHT_LDS_BYTES + wave * WAVE_LDS_BYTES;
    _Float16* qbuf = (_Float16*)(wbase);           // [32][SPITCH] row-major [t][d]
    _Float16* kbuf = (_Float16*)(wbase + 2560);    // [32][SPITCH] row-major [s][d]
    _Float16* vtb  = (_Float16*)(wbase + 5120);    // [32][SPITCH] transposed [d][s]
    float*    weib = (float*)   (wbase + 7680);    // [32][WEIP] f32 [t][s]
    _Float16* attb = (_Float16*)(wbase + 7680);    // alias: [32][SPITCH] [t][s]
    _Float16* outb = (_Float16*)(wbase + 11904);   // [32][WPITCH] concat heads [t][c]

    const float* xb = x + (size_t)b * (T_SEQ * C_DIM);

    // x A-fragments, 2 m-tiles x 4 k-tiles, read x exactly once, cvt to f16
    v16h xa[2][4];
#pragma unroll
    for (int mt = 0; mt < 2; ++mt)
#pragma unroll
      for (int kt = 0; kt < 4; ++kt)
        xa[mt][kt] = a_frag_global_f32(xb + mt * 16 * C_DIM, kt * 32);

#pragma unroll
    for (int h = 0; h < N_HEADS; ++h) {
      const _Float16* wqh = wq_s + h * HDIM * WPITCH;
      const _Float16* wkh = wk_s + h * HDIM * WPITCH;
      const _Float16* wvh = wv_s + h * HDIM * WPITCH;

      // ---- Q = x @ Wq[h]^T ----
      {
        v8f acc[2][2] = {{zero8(), zero8()}, {zero8(), zero8()}};
#pragma unroll
        for (int kt = 0; kt < 4; ++kt) {
          v16h b0 = b_frag_lds(wqh + kt * 32, WPITCH);
          v16h b1 = b_frag_lds(wqh + 16 * WPITCH + kt * 32, WPITCH);
          acc[0][0] = wmma_f16(xa[0][kt], b0, acc[0][0]);
          acc[0][1] = wmma_f16(xa[0][kt], b1, acc[0][1]);
          acc[1][0] = wmma_f16(xa[1][kt], b0, acc[1][0]);
          acc[1][1] = wmma_f16(xa[1][kt], b1, acc[1][1]);
        }
#pragma unroll
        for (int mt = 0; mt < 2; ++mt)
#pragma unroll
          for (int nt = 0; nt < 2; ++nt)
            d_store_rowmajor_h(qbuf + mt * 16 * SPITCH + nt * 16, SPITCH, acc[mt][nt]);
      }
      // ---- K = x @ Wk[h]^T ----
      {
        v8f acc[2][2] = {{zero8(), zero8()}, {zero8(), zero8()}};
#pragma unroll
        for (int kt = 0; kt < 4; ++kt) {
          v16h b0 = b_frag_lds(wkh + kt * 32, WPITCH);
          v16h b1 = b_frag_lds(wkh + 16 * WPITCH + kt * 32, WPITCH);
          acc[0][0] = wmma_f16(xa[0][kt], b0, acc[0][0]);
          acc[0][1] = wmma_f16(xa[0][kt], b1, acc[0][1]);
          acc[1][0] = wmma_f16(xa[1][kt], b0, acc[1][0]);
          acc[1][1] = wmma_f16(xa[1][kt], b1, acc[1][1]);
        }
#pragma unroll
        for (int mt = 0; mt < 2; ++mt)
#pragma unroll
          for (int nt = 0; nt < 2; ++nt)
            d_store_rowmajor_h(kbuf + mt * 16 * SPITCH + nt * 16, SPITCH, acc[mt][nt]);
      }
      // ---- V = x @ Wv[h]^T  (stored transposed: vtb[d][s]) ----
      {
        v8f acc[2][2] = {{zero8(), zero8()}, {zero8(), zero8()}};
#pragma unroll
        for (int kt = 0; kt < 4; ++kt) {
          v16h b0 = b_frag_lds(wvh + kt * 32, WPITCH);
          v16h b1 = b_frag_lds(wvh + 16 * WPITCH + kt * 32, WPITCH);
          acc[0][0] = wmma_f16(xa[0][kt], b0, acc[0][0]);
          acc[0][1] = wmma_f16(xa[0][kt], b1, acc[0][1]);
          acc[1][0] = wmma_f16(xa[1][kt], b0, acc[1][0]);
          acc[1][1] = wmma_f16(xa[1][kt], b1, acc[1][1]);
        }
#pragma unroll
        for (int mt = 0; mt < 2; ++mt)
#pragma unroll
          for (int nt = 0; nt < 2; ++nt)   // tile rows m=s, cols n=d -> vtb[d][s]
            d_store_colmajor_h(vtb + nt * 16 * SPITCH + mt * 16, SPITCH, acc[mt][nt]);
      }
      lds_order();

      // ---- wei = Q @ K^T  (K=32, one k step) ----
      {
        v16h a0 = a_frag_lds(qbuf, SPITCH, 0);
        v16h a1 = a_frag_lds(qbuf + 16 * SPITCH, SPITCH, 0);
        v16h b0 = b_frag_lds(kbuf, SPITCH);              // B[d][s]=k[s][d]
        v16h b1 = b_frag_lds(kbuf + 16 * SPITCH, SPITCH);
        v8f w00 = wmma_f16(a0, b0, zero8());
        v8f w01 = wmma_f16(a0, b1, zero8());
        v8f w10 = wmma_f16(a1, b0, zero8());
        v8f w11 = wmma_f16(a1, b1, zero8());
        d_store_rowmajor_f(weib, WEIP, w00);
        d_store_rowmajor_f(weib + 16, WEIP, w01);
        d_store_rowmajor_f(weib + 16 * WEIP, WEIP, w10);
        d_store_rowmajor_f(weib + 16 * WEIP + 16, WEIP, w11);
      }
      lds_order();

      // ---- causal softmax: lane t owns row t (mask -> *sqrt(C) -> softmax) ----
      {
        const int t = lane;
        float row[T_SEQ];
#pragma unroll
        for (int s = 0; s < T_SEQ; ++s) row[s] = weib[t * WEIP + s];
        float mx = -3.0e38f;
#pragma unroll
        for (int s = 0; s < T_SEQ; ++s) {
          float l = row[s] * ATT_SCALE;
          l = (s <= t) ? l : -3.0e38f;
          row[s] = l;
          mx = fmaxf(mx, l);
        }
        float sum = 0.f;
#pragma unroll
        for (int s = 0; s < T_SEQ; ++s) {
          float p = __expf(row[s] - mx);
          p = (s <= t) ? p : 0.f;
          row[s] = p;
          sum += p;
        }
        float inv = 1.f / sum;
#pragma unroll
        for (int g = 0; g < 4; ++g) {
          v8h hh;
#pragma unroll
          for (int i = 0; i < 8; ++i) hh[i] = (_Float16)(row[g * 8 + i] * inv);
          *(v8h*)(attb + t * SPITCH + g * 8) = hh;
        }
      }
      lds_order();

      // ---- out_h = att @ V ----
      {
        v16h a0 = a_frag_lds(attb, SPITCH, 0);
        v16h a1 = a_frag_lds(attb + 16 * SPITCH, SPITCH, 0);
        v16h b0 = b_frag_lds(vtb, SPITCH);               // B[s][d]=v[s][d] via vtb[d][s]
        v16h b1 = b_frag_lds(vtb + 16 * SPITCH, SPITCH);
        v8f o00 = wmma_f16(a0, b0, zero8());
        v8f o01 = wmma_f16(a0, b1, zero8());
        v8f o10 = wmma_f16(a1, b0, zero8());
        v8f o11 = wmma_f16(a1, b1, zero8());
        _Float16* ob = outb + h * HDIM;
        d_store_rowmajor_h(ob, WPITCH, o00);
        d_store_rowmajor_h(ob + 16, WPITCH, o01);
        d_store_rowmajor_h(ob + 16 * WPITCH, WPITCH, o10);
        d_store_rowmajor_h(ob + 16 * WPITCH + 16, WPITCH, o11);
      }
      lds_order();
    } // heads

    // ---- final projection: y = out @ Wp^T + bp ----
    v16h oa[2][4];
#pragma unroll
    for (int mt = 0; mt < 2; ++mt)
#pragma unroll
      for (int kt = 0; kt < 4; ++kt)
        oa[mt][kt] = a_frag_lds(outb + mt * 16 * WPITCH, WPITCH, kt * 32);

    float* yb = out + (size_t)b * (T_SEQ * C_DIM);
#pragma unroll
    for (int nt = 0; nt < 8; ++nt) {
      v8f acc0 = zero8();
      v8f acc1 = zero8();
#pragma unroll
      for (int kt = 0; kt < 4; ++kt) {
        v16h bb = b_frag_lds(wp_s + nt * 16 * WPITCH + kt * 32, WPITCH);
        acc0 = wmma_f16(oa[0][kt], bb, acc0);
        acc1 = wmma_f16(oa[1][kt], bb, acc1);
      }
      const int n  = nt * 16 + (lane & 15);
      const int mb = (lane >> 4) << 3;
      const float bias = bp[n];
#pragma unroll
      for (int i = 0; i < 8; ++i) {
        yb[(mb + i) * C_DIM + n]      = acc0[i] + bias;
        yb[(16 + mb + i) * C_DIM + n] = acc1[i] + bias;
      }
    }
  }
}

extern "C" void kernel_launch(void* const* d_in, const int* in_sizes, int n_in,
                              void* d_out, int out_size, void* d_ws, size_t ws_size,
                              hipStream_t stream) {
  (void)n_in; (void)out_size; (void)ws_size;
  const float* x  = (const float*)d_in[0];
  const float* Wq = (const float*)d_in[1];
  const float* Wk = (const float*)d_in[2];
  const float* Wv = (const float*)d_in[3];
  const float* Wp = (const float*)d_in[4];
  const float* bp = (const float*)d_in[5];
  float* out = (float*)d_out;
  _Float16* ws = (_Float16*)d_ws;   // 139264 bytes used

  const int nB = in_sizes[0] / (T_SEQ * C_DIM);          // 16384
  const int blocks = (nB + WAVES_PER_BLOCK - 1) / WAVES_PER_BLOCK;

  // stage 1: weights f32 -> padded f16 image in workspace
  cvt_weights_kernel<<<64, 256, 0, stream>>>(Wq, Wk, Wv, Wp, ws);

  // stage 2: attention
  (void)hipFuncSetAttribute((const void*)mha_wmma_kernel,
                            hipFuncAttributeMaxDynamicSharedMemorySize,
                            TOTAL_LDS_BYTES);
  mha_wmma_kernel<<<blocks, 256, TOTAL_LDS_BYTES, stream>>>(x, ws, bp, out, nB);
}
